// FastAttention_14474039787701
// MI455X (gfx1250) — matrix operand order repaired and checked by reading, not compiled, not run on hardware
//
#include <hip/hip_runtime.h>
#include <hip/hip_bf16.h>
#include <math.h>
#include <stdint.h>

// ---------------- problem constants (mirror reference) ----------------
#define B_    2
#define L_    2048
#define NT    (B_ * L_)     // 4096 tokens
#define DM    512
#define DK    64
#define NH    8
#define RK    32
#define RF    128
#define KMAX  64

typedef __attribute__((ext_vector_type(16))) __bf16 v16bf;
typedef __attribute__((ext_vector_type(8)))  float  v8f;

union BF16x16 { v16bf v; unsigned short u[16]; };

// fp32 -> bf16 round-to-nearest-even (scalar fallback)
__device__ __forceinline__ unsigned short f2b(float f) {
  unsigned int u = __float_as_uint(f);
  unsigned int r = 0x7FFFu + ((u >> 16) & 1u);
  return (unsigned short)((u + r) >> 16);
}

// pack two fp32 -> one dword of two bf16 (lo in [15:0], hi in [31:16])
#if __has_builtin(__builtin_amdgcn_cvt_pk_bf16_f32)
typedef __attribute__((ext_vector_type(2))) __bf16 v2bf;
__device__ __forceinline__ unsigned int pack2(float lo, float hi) {
  union { v2bf v; unsigned int u; } cv;
  cv.v = __builtin_amdgcn_cvt_pk_bf16_f32(lo, hi);
  return cv.u;
}
#else
__device__ __forceinline__ unsigned int pack2(float lo, float hi) {
  return (unsigned int)f2b(lo) | ((unsigned int)f2b(hi) << 16);
}
#endif

// =====================================================================
// Fragment-order LDS staging (ISA 7.12.2 layouts inverted).
// Each lane's 16 bf16 operand values are stored contiguously (32 B) ->
// fragment load = two ds_load_b128. Staging does load-all-then-convert-all
// so global b128 loads issue as one clause before any wait.
// =====================================================================

// A tile: 128 rows x 32 k.  (m,k): lane = m + 16*((k>>3)&1), elem=(k&7)+8*(k>>4)
__device__ __forceinline__ void stage_A(const float* __restrict__ A, int lda,
                                        int row0, int k0,
                                        unsigned short* __restrict__ As) {
  float4 v[4];
  int r[4], c4[4];
#pragma unroll
  for (int t = 0; t < 4; ++t) {
    int idx = threadIdx.x + t * 256;
    r[t]  = idx >> 3;
    c4[t] = (idx & 7) << 2;
    v[t] = *(const float4*)(A + (size_t)(row0 + r[t]) * lda + k0 + c4[t]);
  }
#pragma unroll
  for (int t = 0; t < 4; ++t) {
    int wt = r[t] >> 4, m = r[t] & 15;
    int lane = m + (((c4[t] >> 3) & 1) << 4);
    int e0 = (c4[t] & 7) + ((c4[t] >> 4) << 3);          // multiple of 4
    unsigned long long pk = (unsigned long long)pack2(v[t].x, v[t].y) |
                            ((unsigned long long)pack2(v[t].z, v[t].w) << 32);
    *(unsigned long long*)(As + ((wt * 32 + lane) * 16 + e0)) = pk;
  }
}

// B tile: 32 k x N cols.  (k,n): lane = (n&15) + 16*((k>>3)&1), elem = (k&7)+8*(k>>4)
template <int N>
__device__ __forceinline__ void stage_B(const float* __restrict__ W, int ldw,
                                        int k0, int col0,
                                        unsigned short* __restrict__ Bs) {
  constexpr int T = (16 * N) / 256;
  float lo[T], hi[T];
  int n[T], k[T];
#pragma unroll
  for (int t = 0; t < T; ++t) {
    int idx = threadIdx.x + t * 256;
    n[t] = idx & (N - 1);
    k[t] = (idx / N) << 1;
    lo[t] = W[(size_t)(k0 + k[t]) * ldw + col0 + n[t]];
    hi[t] = W[(size_t)(k0 + k[t] + 1) * ldw + col0 + n[t]];
  }
#pragma unroll
  for (int t = 0; t < T; ++t) {
    unsigned int pk = pack2(lo[t], hi[t]);
    int tile = n[t] >> 4, nl = n[t] & 15;
    int lane = nl + (((k[t] >> 3) & 1) << 4);
    *(unsigned int*)(Bs + ((tile * 32 + lane) * 16 + ((k[t] & 7) + ((k[t] >> 4) << 3)))) = pk;
  }
}

// contiguous 32 B per lane -> two ds_load_b128
__device__ __forceinline__ v16bf load_frag(const unsigned short* __restrict__ s) {
  int lane = threadIdx.x & 31;
  const uint4* p = (const uint4*)(s + (size_t)lane * 16);
  BF16x16 f;
  *(uint4*)(f.u)     = p[0];
  *(uint4*)(f.u + 8) = p[1];
  return f.v;
}

// ================= K0: per-head matrices ==========================
// M_h = W_UQ^T @ W_UK  (Qp = q_down @ M_h);  G stacked = (u_v v_v) @ W_out_h
__global__ __launch_bounds__(256) void k_mats(
    const float* __restrict__ u_q, const float* __restrict__ v_q,
    const float* __restrict__ u_k, const float* __restrict__ v_k,
    const float* __restrict__ u_v, const float* __restrict__ v_v,
    const float* __restrict__ W_out, float* __restrict__ M,
    float* __restrict__ G, float* __restrict__ Tws) {
  int h = blockIdx.x;
  __shared__ float WUQ[DK * DK];
  __shared__ float WUK[DK * DK];
  const float* uq = u_q + h * DK * RK; const float* vq = v_q + h * RK * DK;
  const float* uk = u_k + h * DK * RK; const float* vk = v_k + h * RK * DK;
  for (int idx = threadIdx.x; idx < DK * DK; idx += blockDim.x) {
    int i = idx >> 6, j = idx & 63;
    float s1 = 0.f, s2 = 0.f;
    for (int r = 0; r < RK; ++r) {
      s1 += uq[i * RK + r] * vq[r * DK + j];
      s2 += uk[i * RK + r] * vk[r * DK + j];
    }
    WUQ[idx] = s1; WUK[idx] = s2;
  }
  __syncthreads();
  for (int idx = threadIdx.x; idx < DK * DK; idx += blockDim.x) {
    int i = idx >> 6, j = idx & 63;
    float s = 0.f;
    for (int k = 0; k < DK; ++k) s += WUQ[k * DK + i] * WUK[k * DK + j];
    M[h * DK * DK + idx] = s;
  }
  // T = v_v[h] (32x512) @ W_out_h (512x512)  -> ws
  const float* vv = v_v + h * RK * DM;
  float* T = Tws + h * RK * DM;
  for (int idx = threadIdx.x; idx < RK * DM; idx += blockDim.x) {
    int r = idx >> 9, n = idx & 511;
    float s = 0.f;
    for (int m = 0; m < DM; ++m) s += vv[r * DM + m] * W_out[(h * DM + m) * DM + n];
    T[idx] = s;
  }
  __syncthreads();
  // G_h = u_v[h] (64x32) @ T (32x512), stacked at rows h*64..h*64+63
  const float* uv = u_v + h * DK * RK;
  for (int idx = threadIdx.x; idx < DK * DM; idx += blockDim.x) {
    int i = idx >> 9, n = idx & 511;
    float s = 0.f;
    for (int r = 0; r < RK; ++r) s += uv[i * RK + r] * T[r * DM + n];
    G[(h * DK + i) * DM + n] = s;
  }
}

// ================= K1: down projections (WMMA) =====================
// grid(32, 3) block 256: 128 rows x 64 cols per block, K=512
__global__ __launch_bounds__(256) void k_down_proj(
    const float* __restrict__ query, const float* __restrict__ key,
    const float* __restrict__ value, const float* __restrict__ Wq,
    const float* __restrict__ bq, const float* __restrict__ Wkv,
    const float* __restrict__ bkv, float* __restrict__ q_down,
    float* __restrict__ k_down, float* __restrict__ v_down) {
  const float *A, *W, *bias; float* out;
  int z = blockIdx.y;
  if (z == 0)      { A = query; W = Wq;  bias = bq;  out = q_down; }
  else if (z == 1) { A = key;   W = Wkv; bias = bkv; out = k_down; }
  else             { A = value; W = Wkv; bias = bkv; out = v_down; }
  int row0 = blockIdx.x * 128;
  __shared__ __align__(16) unsigned short As[8 * 32 * 16];
  __shared__ __align__(16) unsigned short Bs[4 * 32 * 16];
  int wave = threadIdx.x >> 5;
  v8f acc[4] = {};
  for (int k0 = 0; k0 < DM; k0 += 32) {
    stage_A(A, DM, row0, k0, As);
    if (k0 + 32 < DM)
      __builtin_prefetch(&A[(row0 + (threadIdx.x >> 1)) * DM + k0 + 32], 0, 1);
    stage_B<64>(W, DK, k0, 0, Bs);
    __syncthreads();
    v16bf a = load_frag(As + wave * 32 * 16);
#pragma unroll
    for (int nt = 0; nt < 4; ++nt) {
      v16bf b = load_frag(Bs + nt * 32 * 16);
      acc[nt] = __builtin_amdgcn_wmma_f32_16x16x32_bf16(
          false, a, false, b, (short)0, acc[nt], false, false);
    }
    __syncthreads();
  }
  int lane = threadIdx.x & 31;
  int n0 = lane & 15, mb = (lane >> 4) << 3;
#pragma unroll
  for (int nt = 0; nt < 4; ++nt) {
    int col = nt * 16 + n0;
    float bv = bias[col];
#pragma unroll
    for (int r = 0; r < 8; ++r)
      out[(row0 + wave * 16 + mb + r) * DK + col] = acc[nt][r] + bv;
  }
}

// ================= K2: Qp = q_down @ M_h (WMMA) ====================
// grid(32, NH) block 256: 128 rows x 64 cols, K=64
__global__ __launch_bounds__(256) void k_qp(
    const float* __restrict__ q_down, const float* __restrict__ M,
    float* __restrict__ Qp) {
  int h = blockIdx.y;
  int row0 = blockIdx.x * 128;
  const float* Bm = M + h * DK * DK;
  float* out = Qp + (size_t)h * NT * DK;
  __shared__ __align__(16) unsigned short As[8 * 32 * 16];
  __shared__ __align__(16) unsigned short Bs[4 * 32 * 16];
  int wave = threadIdx.x >> 5;
  v8f acc[4] = {};
  for (int k0 = 0; k0 < DK; k0 += 32) {
    stage_A(q_down, DK, row0, k0, As);
    stage_B<64>(Bm, DK, k0, 0, Bs);
    __syncthreads();
    v16bf a = load_frag(As + wave * 32 * 16);
#pragma unroll
    for (int nt = 0; nt < 4; ++nt) {
      v16bf b = load_frag(Bs + nt * 32 * 16);
      acc[nt] = __builtin_amdgcn_wmma_f32_16x16x32_bf16(
          false, a, false, b, (short)0, acc[nt], false, false);
    }
    __syncthreads();
  }
  int lane = threadIdx.x & 31;
  int n0 = lane & 15, mb = (lane >> 4) << 3;
#pragma unroll
  for (int nt = 0; nt < 4; ++nt) {
    int col = nt * 16 + n0;
#pragma unroll
    for (int r = 0; r < 8; ++r)
      out[(row0 + wave * 16 + mb + r) * DK + col] = acc[nt][r];
  }
}

// ================= K3: sign-bit codes via wave32 ballot ============
// one wave per (source, token): 64 sign bits = two __ballot's.
__global__ __launch_bounds__(256) void k_codes(
    const float* __restrict__ Qp, const float* __restrict__ k_down,
    unsigned long long* __restrict__ qcodes,
    unsigned long long* __restrict__ kcodes) {
  int wave = threadIdx.x >> 5, lane = threadIdx.x & 31;
  int gw = blockIdx.x * 8 + wave;
  int z = gw / NT, tok = gw % NT;
  const float* src = (z < NH) ? (Qp + ((size_t)z * NT + tok) * DK)
                              : (k_down + (size_t)tok * DK);
  unsigned long long m0 = __ballot(src[lane] > 0.0f);
  unsigned long long m1 = __ballot(src[lane + 32] > 0.0f);
  unsigned long long code = (m0 & 0xFFFFFFFFull) | (m1 << 32);
  if (lane == 0) {
    if (z < NH) qcodes[(size_t)z * NT + tok] = code;
    else        kcodes[tok] = code;
  }
}

// ================= K4: phi = cos(X@omega + b)*scale (WMMA) =========
// grid(32, NH, 2) block 256: 128 rows x 128 cols, K=64
__global__ __launch_bounds__(256) void k_phi(
    const float* __restrict__ Qp, const float* __restrict__ k_down,
    const float* __restrict__ omega, const float* __restrict__ rff_bias,
    float* __restrict__ phi_q, float* __restrict__ phi_k,
    float scale_q, float scale_k) {
  int h = blockIdx.y;
  int z = blockIdx.z;
  const float* A = z ? (k_down) : (Qp + (size_t)h * NT * DK);
  const float* Bm = omega + (size_t)h * DK * RF;
  const float* bias = rff_bias + h * RF;
  float* out = (z ? phi_k : phi_q) + (size_t)h * NT * RF;
  float scale = z ? scale_k : scale_q;
  int row0 = blockIdx.x * 128;
  __shared__ __align__(16) unsigned short As[8 * 32 * 16];
  __shared__ __align__(16) unsigned short Bs[8 * 32 * 16];
  int wave = threadIdx.x >> 5;
  v8f acc[8] = {};
  for (int k0 = 0; k0 < DK; k0 += 32) {
    stage_A(A, DK, row0, k0, As);
    stage_B<128>(Bm, RF, k0, 0, Bs);
    __syncthreads();
    v16bf a = load_frag(As + wave * 32 * 16);
#pragma unroll
    for (int nt = 0; nt < 8; ++nt) {
      v16bf b = load_frag(Bs + nt * 32 * 16);
      acc[nt] = __builtin_amdgcn_wmma_f32_16x16x32_bf16(
          false, a, false, b, (short)0, acc[nt], false, false);
    }
    __syncthreads();
  }
  int lane = threadIdx.x & 31;
  int n0 = lane & 15, mb = (lane >> 4) << 3;
#pragma unroll
  for (int nt = 0; nt < 8; ++nt) {
    int col = nt * 16 + n0;
    float bv = bias[col];
#pragma unroll
    for (int r = 0; r < 8; ++r)
      out[(row0 + wave * 16 + mb + r) * RF + col] = cosf(acc[nt][r] + bv) * scale;
  }
}

// ================= K5: candidate search ============================
// one wave32 per (head, token): per-group first-64 cumsum cap via ballot
// prefix counts; union; first-64 ascending with uniform early exit.
__global__ __launch_bounds__(256) void k_cand(
    const unsigned long long* __restrict__ qcodes,
    const unsigned long long* __restrict__ kcodes,
    unsigned short* __restrict__ cand, int* __restrict__ cnt) {
  int gwave = (blockIdx.x * 256 + threadIdx.x) >> 5;   // 0 .. NH*NT-1
  int lane = threadIdx.x & 31;
  int h = gwave / NT, tok = gwave % NT, b = tok / L_;
  unsigned long long qc = qcodes[(size_t)h * NT + tok];
  const unsigned long long* kc = kcodes + (size_t)b * L_;
  unsigned short* cd = cand + ((size_t)h * NT + tok) * KMAX;
  unsigned lmask_lt = (1u << lane) - 1u;
  int grp[8] = {0, 0, 0, 0, 0, 0, 0, 0};
  int nc = 0;
  for (int j0 = 0; j0 < L_ && nc < KMAX; j0 += 32) {
    int j = j0 + lane;
    unsigned long long x = qc ^ kc[j];
    unsigned keep = 0u;
#pragma unroll
    for (int g = 0; g < 8; ++g) {
      bool mg = ((x >> (8 * g)) & 0xFFull) == 0ull;
      unsigned mk = (unsigned)__ballot(mg);
      if (mg && (grp[g] + __popc(mk & lmask_lt)) < KMAX) keep = 1u;
      grp[g] += __popc(mk);
    }
    unsigned um = (unsigned)__ballot(keep != 0u);
    int pos = nc + __popc(um & lmask_lt);
    if (keep && pos < KMAX) cd[pos] = (unsigned short)j;
    nc += __popc(um);
  }
  if (nc > KMAX) nc = KMAX;
  for (int p = nc + lane; p < KMAX; p += 32) cd[p] = 0;
  if (lane == 0) cnt[(size_t)h * NT + tok] = nc;
}

// ================= K6: gathered scores + softmax + ctx =============
// one wave32 per (head, token); block = 8 waves.
// phi_q row (512 B/wave) staged global->LDS with the CDNA5 async path:
// GLOBAL_LOAD_ASYNC_TO_LDS_B128 (ASYNCcnt), no VGPR roundtrip.
__global__ __launch_bounds__(256) void k_attn(
    const float* __restrict__ phi_q, const float* __restrict__ phi_k,
    const float* __restrict__ v_down, const unsigned short* __restrict__ cand,
    const int* __restrict__ cnt, float* __restrict__ Cc) {
  __shared__ __align__(16) float pq[8][RF];
  __shared__ float wsm[8][KMAX];
  __shared__ unsigned short cds[8][KMAX];
  int wave = threadIdx.x >> 5, lane = threadIdx.x & 31;
  int gwave = blockIdx.x * 8 + wave;
  int h = gwave / NT, tok = gwave % NT, b = tok / L_;
  const float* pqrow = phi_q + ((size_t)h * NT + tok) * RF;
  // async copy: 32 lanes x 16 B = full 512 B row directly into LDS
  {
    unsigned ldsoff = (unsigned)(uintptr_t)(&pq[wave][0]) + (unsigned)(lane * 16);
    unsigned long long gaddr = (unsigned long long)(uintptr_t)(pqrow + lane * 4);
    asm volatile("global_load_async_to_lds_b128 %0, %1, off"
                 :: "v"(ldsoff), "v"(gaddr)
                 : "memory");
  }
  int n = cnt[(size_t)h * NT + tok];
  const unsigned short* cd = cand + ((size_t)h * NT + tok) * KMAX;
  cds[wave][lane] = cd[lane];
  cds[wave][lane + 32] = cd[lane + 32];
  asm volatile("s_wait_asynccnt 0x0" ::: "memory");
  __syncthreads();
  const float* pkbase = phi_k + (size_t)h * NT * RF + (size_t)b * L_ * RF;
  const float4* pq4 = (const float4*)(&pq[wave][0]);
  float s[2];
#pragma unroll
  for (int half = 0; half < 2; ++half) {
    int c = lane + 32 * half;
    float acc = -1e30f;
    if (c < n) {
      const float4* pk4 =
          (const float4*)(pkbase + (size_t)cds[wave][c] * RF);
      float a = 0.f;
#pragma unroll 8
      for (int r = 0; r < RF / 4; ++r) {
        float4 x = pk4[r], y = pq4[r];
        a += x.x * y.x + x.y * y.y + x.z * y.z + x.w * y.w;
      }
      acc = a;  // 1/sqrt(RF) folded into phi_q scale
    }
    s[half] = acc;
  }
  float m = fmaxf(s[0], s[1]);
  for (int off = 16; off; off >>= 1) m = fmaxf(m, __shfl_xor(m, off, 32));
  float e0 = __expf(s[0] - m), e1 = __expf(s[1] - m);
  float sum = e0 + e1;
  for (int off = 16; off; off >>= 1) sum += __shfl_xor(sum, off, 32);
  float inv = 1.0f / sum;
  wsm[wave][lane]      = (lane < n)      ? e0 * inv : 0.f;
  wsm[wave][lane + 32] = (lane + 32 < n) ? e1 * inv : 0.f;
  __syncthreads();
  // ctx: lane owns dims {lane, lane+32}; coalesced v_down gathers
  float c0 = 0.f, c1 = 0.f;
  const float* vb = v_down + (size_t)b * L_ * DK;
  for (int c = 0; c < KMAX; ++c) {
    float w = wsm[wave][c];
    if (w != 0.f) {
      const float* vr = vb + (size_t)cds[wave][c] * DK;
      c0 += w * vr[lane];
      c1 += w * vr[lane + 32];
    }
  }
  float* outrow = Cc + (size_t)tok * DM + h * DK;
  outrow[lane] = c0;
  outrow[lane + 32] = c1;
}

// ================= K7: final fused GEMM out = Cc @ G + b_out =======
// grid(32, 8) block 256: 128 rows x 64 cols, K=512
__global__ __launch_bounds__(256) void k_final(
    const float* __restrict__ Cc, const float* __restrict__ G,
    const float* __restrict__ b_out, float* __restrict__ out) {
  int row0 = blockIdx.x * 128;
  int ncol0 = blockIdx.y * 64;
  __shared__ __align__(16) unsigned short As[8 * 32 * 16];
  __shared__ __align__(16) unsigned short Bs[4 * 32 * 16];
  int wave = threadIdx.x >> 5;
  v8f acc[4] = {};
  for (int k0 = 0; k0 < DM; k0 += 32) {
    stage_A(Cc, DM, row0, k0, As);
    if (k0 + 32 < DM)
      __builtin_prefetch(&Cc[(row0 + (threadIdx.x >> 1)) * DM + k0 + 32], 0, 1);
    stage_B<64>(G, DM, k0, ncol0, Bs);
    __syncthreads();
    v16bf a = load_frag(As + wave * 32 * 16);
#pragma unroll
    for (int nt = 0; nt < 4; ++nt) {
      v16bf b = load_frag(Bs + nt * 32 * 16);
      acc[nt] = __builtin_amdgcn_wmma_f32_16x16x32_bf16(
          false, a, false, b, (short)0, acc[nt], false, false);
    }
    __syncthreads();
  }
  int lane = threadIdx.x & 31;
  int n0 = lane & 15, mb = (lane >> 4) << 3;
#pragma unroll
  for (int nt = 0; nt < 4; ++nt) {
    int col = ncol0 + nt * 16 + n0;
    float bv = b_out[col];
#pragma unroll
    for (int r = 0; r < 8; ++r)
      out[(row0 + wave * 16 + mb + r) * DM + col] = acc[nt][r] + bv;
  }
}

// =====================================================================
extern "C" void kernel_launch(void* const* d_in, const int* in_sizes, int n_in,
                              void* d_out, int out_size, void* d_ws, size_t ws_size,
                              hipStream_t stream) {
  (void)in_sizes; (void)n_in; (void)out_size; (void)ws_size;
  const float* query    = (const float*)d_in[0];
  const float* key      = (const float*)d_in[1];
  const float* value    = (const float*)d_in[2];
  const float* Wq       = (const float*)d_in[3];
  const float* bq       = (const float*)d_in[4];
  const float* Wkv      = (const float*)d_in[5];
  const float* bkv      = (const float*)d_in[6];
  const float* u_q      = (const float*)d_in[7];
  const float* v_q      = (const float*)d_in[8];
  const float* u_k      = (const float*)d_in[9];
  const float* v_k      = (const float*)d_in[10];
  const float* u_v      = (const float*)d_in[11];
  const float* v_v      = (const float*)d_in[12];
  const float* omega    = (const float*)d_in[13];
  const float* rff_bias = (const float*)d_in[14];
  const float* W_out    = (const float*)d_in[15];
  const float* b_out    = (const float*)d_in[16];
  float* out = (float*)d_out;

  char* ws = (char*)d_ws;
  size_t off = 0;
  auto alloc = [&](size_t bytes) -> void* {
    void* p = ws + off;
    off += (bytes + 255) & ~(size_t)255;
    return p;
  };
  float* M      = (float*)alloc((size_t)NH * DK * DK * 4);     // 128 KB
  float* G      = (float*)alloc((size_t)DM * DM * 4);          // 1 MB (stacked)
  float* Tws    = (float*)alloc((size_t)NH * RK * DM * 4);     // 512 KB
  float* q_down = (float*)alloc((size_t)NT * DK * 4);          // 1 MB
  float* k_down = (float*)alloc((size_t)NT * DK * 4);
  float* v_down = (float*)alloc((size_t)NT * DK * 4);
  float* Qp     = (float*)alloc((size_t)NH * NT * DK * 4);     // 8 MB
  unsigned long long* qcodes = (unsigned long long*)alloc((size_t)NH * NT * 8);
  unsigned long long* kcodes = (unsigned long long*)alloc((size_t)NT * 8);
  float* phi_q  = (float*)alloc((size_t)NH * NT * RF * 4);     // 16 MB
  float* phi_k  = (float*)alloc((size_t)NH * NT * RF * 4);     // 16 MB
  unsigned short* cand = (unsigned short*)alloc((size_t)NH * NT * KMAX * 2);
  int* cnt      = (int*)alloc((size_t)NH * NT * 4);
  float* Cc     = (float*)alloc((size_t)NT * DM * 4);          // 8 MB

  const float rff_scale = 0.125f;                  // sqrt(2/128)
  const float inv_sqrt  = 0.08838834764831845f;    // 1/sqrt(128)

  k_mats<<<NH, 256, 0, stream>>>(u_q, v_q, u_k, v_k, u_v, v_v, W_out, M, G, Tws);
  k_down_proj<<<dim3(NT / 128, 3), 256, 0, stream>>>(
      query, key, value, Wq, bq, Wkv, bkv, q_down, k_down, v_down);
  k_qp<<<dim3(NT / 128, NH), 256, 0, stream>>>(q_down, M, Qp);
  k_codes<<<((NH + 1) * NT) / 8, 256, 0, stream>>>(Qp, k_down, qcodes, kcodes);
  k_phi<<<dim3(NT / 128, NH, 2), 256, 0, stream>>>(
      Qp, k_down, omega, rff_bias, phi_q, phi_k, rff_scale * inv_sqrt, rff_scale);
  k_cand<<<(NH * NT) / 8, 256, 0, stream>>>(qcodes, kcodes, cand, cnt);
  k_attn<<<(NH * NT) / 8, 256, 0, stream>>>(phi_q, phi_k, v_down, cand, cnt, Cc);
  k_final<<<dim3(NT / 128, DM / 64), 256, 0, stream>>>(Cc, G, b_out, out);
}
